// Attention_44160853738098
// MI455X (gfx1250) — matrix operand order repaired and checked by reading, not compiled, not run on hardware
//
#include <hip/hip_runtime.h>

typedef __bf16 v16bf __attribute__((ext_vector_type(16)));
typedef float  v8f   __attribute__((ext_vector_type(8)));
typedef unsigned short u16x8 __attribute__((ext_vector_type(8)));
typedef unsigned short u16x4 __attribute__((ext_vector_type(4)));
typedef unsigned int   u32x4 __attribute__((ext_vector_type(4)));
typedef int            i32x4 __attribute__((ext_vector_type(4)));
typedef int            i32x8 __attribute__((ext_vector_type(8)));

static constexpr int S   = 2048;
static constexpr int DIM = 4096;
static constexpr int HD  = 128;
static constexpr int NH  = 32;
static constexpr int NKV = 8;
static constexpr int KVD = NKV * HD;   // 1024

__device__ __forceinline__ unsigned short bf16_bits(float x) {
  __bf16 b = (__bf16)x;
  return __builtin_bit_cast(unsigned short, b);
}

// A-operand fragment (16x32, MxK): lane half 0 -> K {0..7, 16..23}, half 1 -> K {8..15, 24..31}
__device__ __forceinline__ v16bf load_frag_a(const __bf16* base, int ld, int lane) {
  const int m = lane & 15, hf = lane >> 4;
  const __bf16* p = base + (size_t)m * ld + hf * 8;
  union { u16x8 h[2]; v16bf v; } f;
  f.h[0] = *(const u16x8*)(p);
  f.h[1] = *(const u16x8*)(p + 16);
  return f.v;
}

// B-operand fragment (32x16, KxN) from row-major [N][K] source (B^T rows):
// lane half 0 -> K {0..7, 16..23}, half 1 -> K {8..15, 24..31}
__device__ __forceinline__ v16bf load_frag_b(const __bf16* base, int ld, int lane) {
  const int n = lane & 15, hf = lane >> 4;
  const __bf16* p = base + (size_t)n * ld + hf * 8;
  union { u16x8 h[2]; v16bf v; } f;
  f.h[0] = *(const u16x8*)(p);
  f.h[1] = *(const u16x8*)(p + 16);
  return f.v;
}

// ---------------------------------------------------------------------------
// TDM: DMA a [128 rows x 32 bf16] tile (row stride K elems) into LDS with
// hardware padding: after every 16 DWORDs (64B = one 32-elem row) insert
// 4 DWORDs (16B) -> LDS row stride 40 bf16 (matches fragment loaders).
// D# per CDNA5 ISA 8.3/8.4: group0 {count=1, lds_addr, global_addr, type=2},
// group1 {data_size=2B, pad_enable, pad_interval=3(16DW), pad_amount=3(4DW),
//         tensor_dim0=K, tensor_dim1=128, tile_dim0=32, tile_dim1=128,
//         tensor_dim0_stride=K}.
// ---------------------------------------------------------------------------
__device__ __forceinline__ void tdm_load_tile(unsigned lds_off, const __bf16* gptr, int K) {
  unsigned long long ga = (unsigned long long)(size_t)gptr;
  u32x4 g0;
  g0[0] = 1u;                                              // count=1 (valid descriptor)
  g0[1] = lds_off;                                         // lds_addr [63:32]
  g0[2] = (unsigned)ga;                                    // global_addr [95:64]
  g0[3] = (unsigned)((ga >> 32) & 0x1FFFFFFu) | (2u << 30);// global_addr hi | type=2
  i32x8 g1;
  g1[0] = (1 << 16) | (1 << 20) | (3 << 22) | (3 << 25);   // 2B elems, pad 16DW->+4DW
  g1[1] = (int)((K & 0xFFFF) << 16);                       // tensor_dim0 lo16
  g1[2] = (int)(((unsigned)K >> 16) & 0xFFFFu) | (128 << 16); // dim0 hi | tensor_dim1 lo
  g1[3] = (int)(32u << 16);                                // tensor_dim1 hi=0 | tile_dim0=32
  g1[4] = 128;                                             // tile_dim1=128, tile_dim2=0
  g1[5] = K;                                               // tensor_dim0_stride lo32
  g1[6] = 0;                                               // stride hi | dim1_stride lo
  g1[7] = 0;
  i32x4 z4 = {0, 0, 0, 0};
#if defined(__clang_major__) && __clang_major__ >= 23
  i32x8 z8 = {0, 0, 0, 0, 0, 0, 0, 0};
  __builtin_amdgcn_tensor_load_to_lds(g0, g1, z4, z4, z8, 0);
#else
  __builtin_amdgcn_tensor_load_to_lds(g0, g1, z4, z4, 0);
#endif
}

// ---------------------------------------------------------------------------
// C[M,N] = A[M,K] * B[N,K]^T, all-bf16 operands, f32 accumulate/output.
// block 128x128, BK=32, 8 waves (2x4), wave tile 64x32.
// Double-buffered LDS staged by the Tensor Data Mover (wave 0 issues,
// TENSORcnt-synchronized, overlapped with WMMA on the other buffer).
// ---------------------------------------------------------------------------
__global__ __launch_bounds__(256) void gemm_bt_bf16_tdm(const __bf16* __restrict__ A,
                                                        const __bf16* __restrict__ B,
                                                        float* __restrict__ C,
                                                        int M, int N, int K) {
  __shared__ __bf16 smem[2][2][128 * 40];  // [buf][A/B][tile]
  const int tid  = threadIdx.x;
  const int lane = tid & 31;
  const int w    = tid >> 5;
  const int mbase = blockIdx.y * 128, nbase = blockIdx.x * 128;
  const int wm = (w >> 2) * 64, wn = (w & 3) * 32;

  v8f acc[4][2];
#pragma unroll
  for (int i = 0; i < 4; ++i)
#pragma unroll
    for (int j = 0; j < 2; ++j)
#pragma unroll
      for (int v = 0; v < 8; ++v) acc[i][j][v] = 0.0f;

  const __bf16* Ab = A + (size_t)mbase * K;
  const __bf16* Bb = B + (size_t)nbase * K;

  if (w == 0) {
    tdm_load_tile((unsigned)(size_t)&smem[0][0][0], Ab, K);
    tdm_load_tile((unsigned)(size_t)&smem[0][1][0], Bb, K);
    if (32 < K) {
      tdm_load_tile((unsigned)(size_t)&smem[1][0][0], Ab + 32, K);
      tdm_load_tile((unsigned)(size_t)&smem[1][1][0], Bb + 32, K);
    }
  }

  int buf = 0;
  for (int kk = 0; kk < K; kk += 32) {
    const bool has_next = (kk + 32 < K);
    if (w == 0) {
      if (has_next) __builtin_amdgcn_s_wait_tensorcnt((short)2);  // current tile landed
      else          __builtin_amdgcn_s_wait_tensorcnt((short)0);
    }
    __syncthreads();

    const __bf16* As = &smem[buf][0][0];
    const __bf16* Bs = &smem[buf][1][0];
    v16bf a[4], b[2];
#pragma unroll
    for (int i = 0; i < 4; ++i) a[i] = load_frag_a(As + (wm + i * 16) * 40, 40, lane);
#pragma unroll
    for (int j = 0; j < 2; ++j) b[j] = load_frag_b(Bs + (wn + j * 16) * 40, 40, lane);
#pragma unroll
    for (int i = 0; i < 4; ++i)
#pragma unroll
      for (int j = 0; j < 2; ++j)
        acc[i][j] = __builtin_amdgcn_wmma_f32_16x16x32_bf16(
            false, a[i], false, b[j], (short)0, acc[i][j], false, false);

    __syncthreads();  // everyone done reading smem[buf] before DMA reuses it
    if (w == 0 && kk + 64 < K) {
      tdm_load_tile((unsigned)(size_t)&smem[buf][0][0], Ab + kk + 64, K);
      tdm_load_tile((unsigned)(size_t)&smem[buf][1][0], Bb + kk + 64, K);
    }
    buf ^= 1;
  }

  const int nl = lane & 15, hi = lane >> 4;
#pragma unroll
  for (int i = 0; i < 4; ++i)
#pragma unroll
    for (int j = 0; j < 2; ++j)
#pragma unroll
      for (int v = 0; v < 8; ++v)
        C[(size_t)(mbase + wm + i * 16 + v + 8 * hi) * N + (nbase + wn + j * 16 + nl)] =
            acc[i][j][v];
}

// ---------------------------------------------------------------------------
// One-shot f32 -> bf16 conversion (vectorized), removes cvt from GEMM loops.
// ---------------------------------------------------------------------------
__global__ __launch_bounds__(256) void f32_to_bf16_kernel(const float* __restrict__ src,
                                                          __bf16* __restrict__ dst, int n4) {
  int i = blockIdx.x * 256 + threadIdx.x;
  if (i < n4) {
    float4 v = ((const float4*)src)[i];
    u16x4 o;
    o[0] = bf16_bits(v.x); o[1] = bf16_bits(v.y);
    o[2] = bf16_bits(v.z); o[3] = bf16_bits(v.w);
    ((u16x4*)dst)[i] = o;
  }
}

// ---------------------------------------------------------------------------
// RoPE + RMSNorm on Q/K (f32 -> bf16, head-major), V convert to dim-major bf16
// ---------------------------------------------------------------------------
__global__ __launch_bounds__(256) void rope_norm_kernel(const float* __restrict__ Qf,
                                                        const float* __restrict__ Kf,
                                                        const float* __restrict__ Vf,
                                                        const float* __restrict__ fc,
                                                        __bf16* __restrict__ Qb,
                                                        __bf16* __restrict__ Kb,
                                                        __bf16* __restrict__ Vt) {
  const int lane = threadIdx.x & 31;
  const int row  = blockIdx.x * 8 + (threadIdx.x >> 5);
  const int d0   = lane * 4;
  const int NQ = S * NH, NK = S * NKV;

  if (row < NQ + NK) {
    int s, hh;
    const float* src;
    __bf16* dst;
    if (row < NQ) {
      s = row >> 5; hh = row & 31;
      src = Qf + (size_t)s * DIM + hh * HD;
      dst = Qb + ((size_t)hh * S + s) * HD;
    } else {
      int r = row - NQ;
      s = r >> 3; hh = r & 7;
      src = Kf + (size_t)s * KVD + hh * HD;
      dst = Kb + ((size_t)hh * S + s) * HD;
    }
    float4 xv = *(const float4*)(src + d0);
    float4 fv = *(const float4*)(fc + (size_t)s * HD + d0);  // (cos,sin,cos,sin)
    float na0 = xv.x * fv.x - xv.y * fv.y;
    float nb0 = xv.x * fv.y + xv.y * fv.x;
    float na1 = xv.z * fv.z - xv.w * fv.w;
    float nb1 = xv.z * fv.w + xv.w * fv.z;
    float ss = na0 * na0 + nb0 * nb0 + na1 * na1 + nb1 * nb1;
#pragma unroll
    for (int m = 1; m < 32; m <<= 1) ss += __shfl_xor(ss, m, 32);
    float r = rsqrtf(ss * (1.0f / 128.0f) + 1e-5f);
    u16x4 o;
    o[0] = bf16_bits(na0 * r); o[1] = bf16_bits(nb0 * r);
    o[2] = bf16_bits(na1 * r); o[3] = bf16_bits(nb1 * r);
    *(u16x4*)(dst + d0) = o;
  } else {
    int r = row - NQ - NK;
    int s = r >> 3, hh = r & 7;
    const float* src = Vf + (size_t)s * KVD + hh * HD;
    float4 xv = *(const float4*)(src + d0);
    __bf16* dst = Vt + (size_t)hh * HD * S + s;  // [kvh][d][s]
    dst[(size_t)(d0 + 0) * S] = (__bf16)xv.x;
    dst[(size_t)(d0 + 1) * S] = (__bf16)xv.y;
    dst[(size_t)(d0 + 2) * S] = (__bf16)xv.z;
    dst[(size_t)(d0 + 3) * S] = (__bf16)xv.w;
  }
}

// ---------------------------------------------------------------------------
// Causal flash attention: grid (S/64, NH), 4 waves/WG, wave owns 16 q-rows.
// ---------------------------------------------------------------------------
__global__ __launch_bounds__(128) void attn_kernel(const __bf16* __restrict__ Qb,
                                                   const __bf16* __restrict__ Kb,
                                                   const __bf16* __restrict__ Vt,
                                                   __bf16* __restrict__ Obf) {
  __shared__ __bf16 Plds[4][16 * 40];
  const int lane = threadIdx.x & 31, w = threadIdx.x >> 5;
  const int h = blockIdx.y, kvh = h >> 2;  // n_rep = 4
  const int qw = blockIdx.x * 64 + w * 16;
  const __bf16* Qh = Qb + (size_t)h   * S * HD;
  const __bf16* Kh = Kb + (size_t)kvh * S * HD;
  const __bf16* Vh = Vt + (size_t)kvh * HD * S;
  __bf16* Pw = &Plds[w][0];

  v16bf aq[4];
#pragma unroll
  for (int dc = 0; dc < 4; ++dc)
    aq[dc] = load_frag_a(Qh + (size_t)qw * HD + dc * 32, HD, lane);

  v8f acc[8];
  float mi[8], li[8];
#pragma unroll
  for (int v = 0; v < 8; ++v) { mi[v] = -3.0e38f; li[v] = 0.0f; }
#pragma unroll
  for (int g = 0; g < 8; ++g)
#pragma unroll
    for (int v = 0; v < 8; ++v) acc[g][v] = 0.0f;

  const int nl = lane & 15, hi = lane >> 4;
  const float scale = 0.08838834764831845f;  // 1/sqrt(128)
  const int kend = qw + 16;

  for (int kb = 0; kb < kend; kb += 32) {
    float sv[2][8];
#pragma unroll
    for (int b = 0; b < 2; ++b) {
      v8f s;
#pragma unroll
      for (int v = 0; v < 8; ++v) s[v] = 0.0f;
#pragma unroll
      for (int dc = 0; dc < 4; ++dc) {
        v16bf kf = load_frag_b(Kh + (size_t)(kb + b * 16) * HD + dc * 32, HD, lane);
        s = __builtin_amdgcn_wmma_f32_16x16x32_bf16(false, aq[dc], false, kf,
                                                    (short)0, s, false, false);
      }
#pragma unroll
      for (int v = 0; v < 8; ++v) {
        int kn = kb + b * 16 + nl;
        int qm = qw + v + 8 * hi;
        sv[b][v] = (kn <= qm) ? s[v] * scale : -3.0e38f;  // causal mask
      }
    }
#pragma unroll
    for (int v = 0; v < 8; ++v) {
      float rm = fmaxf(sv[0][v], sv[1][v]);
#pragma unroll
      for (int msk = 1; msk < 16; msk <<= 1) rm = fmaxf(rm, __shfl_xor(rm, msk, 32));
      float mnew  = fmaxf(mi[v], rm);
      float alpha = __expf(mi[v] - mnew);
      float p0 = __expf(sv[0][v] - mnew);
      float p1 = __expf(sv[1][v] - mnew);
      float rs = p0 + p1;
#pragma unroll
      for (int msk = 1; msk < 16; msk <<= 1) rs += __shfl_xor(rs, msk, 32);
      li[v] = li[v] * alpha + rs;
      mi[v] = mnew;
#pragma unroll
      for (int g = 0; g < 8; ++g) acc[g][v] *= alpha;
      int m = v + 8 * hi;
      Pw[m * 40 + nl]      = (__bf16)p0;   // C-layout -> A-layout via LDS
      Pw[m * 40 + 16 + nl] = (__bf16)p1;
    }
    asm volatile("s_wait_dscnt 0x0" ::: "memory");  // own-wave LDS RAW fence
    v16bf pf = load_frag_a(Pw, 40, lane);           // P as 16x32 A-operand
#pragma unroll
    for (int g = 0; g < 8; ++g) {
      v16bf vf = load_frag_b(Vh + (size_t)(g * 16) * S + kb, S, lane);
      acc[g] = __builtin_amdgcn_wmma_f32_16x16x32_bf16(false, pf, false, vf,
                                                       (short)0, acc[g], false, false);
    }
  }

#pragma unroll
  for (int v = 0; v < 8; ++v) {
    float inv = 1.0f / li[v];
    int m = qw + v + 8 * hi;
#pragma unroll
    for (int g = 0; g < 8; ++g)
      Obf[(size_t)m * DIM + h * HD + g * 16 + nl] = (__bf16)(acc[g][v] * inv);
  }
}

// ---------------------------------------------------------------------------
extern "C" void kernel_launch(void* const* d_in, const int* in_sizes, int n_in,
                              void* d_out, int out_size, void* d_ws, size_t ws_size,
                              hipStream_t stream) {
  (void)in_sizes; (void)n_in; (void)out_size; (void)ws_size;
  const float* x  = (const float*)d_in[0];
  const float* wq = (const float*)d_in[1];
  const float* wk = (const float*)d_in[2];
  const float* wv = (const float*)d_in[3];
  const float* wo = (const float*)d_in[4];
  const float* fc = (const float*)d_in[5];
  float* out = (float*)d_out;

  __bf16* xb  = (__bf16*)d_ws;                       // [S, DIM]
  __bf16* wqb = xb  + (size_t)S * DIM;               // [DIM, DIM]
  __bf16* wkb = wqb + (size_t)DIM * DIM;             // [KVD, DIM]
  __bf16* wvb = wkb + (size_t)KVD * DIM;             // [KVD, DIM]
  __bf16* wob = wvb + (size_t)KVD * DIM;             // [DIM, DIM]
  float*  Qf  = (float*)(wob + (size_t)DIM * DIM);   // [S, DIM]
  float*  Kf  = Qf + (size_t)S * DIM;                // [S, KVD]
  float*  Vf  = Kf + (size_t)S * KVD;                // [S, KVD]
  __bf16* Qb  = (__bf16*)(Vf + (size_t)S * KVD);     // [NH][S][HD]
  __bf16* Kb  = Qb + (size_t)S * DIM;                // [NKV][S][HD]
  __bf16* Vt  = Kb + (size_t)S * KVD;                // [NKV][HD][S]
  __bf16* Ob  = xb;                                  // alias: xb dead after V GEMM

  auto cvt = [&](const float* s, __bf16* d, size_t n) {
    int n4 = (int)(n / 4);
    f32_to_bf16_kernel<<<(n4 + 255) / 256, 256, 0, stream>>>(s, d, n4);
  };
  cvt(x,  xb,  (size_t)S * DIM);
  cvt(wq, wqb, (size_t)DIM * DIM);
  cvt(wk, wkb, (size_t)KVD * DIM);
  cvt(wv, wvb, (size_t)KVD * DIM);
  cvt(wo, wob, (size_t)DIM * DIM);

  gemm_bt_bf16_tdm<<<dim3(DIM / 128, S / 128), 256, 0, stream>>>(xb, wqb, Qf, S, DIM, DIM);
  gemm_bt_bf16_tdm<<<dim3(KVD / 128, S / 128), 256, 0, stream>>>(xb, wkb, Kf, S, KVD, DIM);
  gemm_bt_bf16_tdm<<<dim3(KVD / 128, S / 128), 256, 0, stream>>>(xb, wvb, Vf, S, KVD, DIM);
  rope_norm_kernel<<<(S * (NH + 2 * NKV)) / 8, 256, 0, stream>>>(Qf, Kf, Vf, fc, Qb, Kb, Vt);
  attn_kernel<<<dim3(S / 64, NH), 128, 0, stream>>>(Qb, Kb, Vt, Ob);
  gemm_bt_bf16_tdm<<<dim3(DIM / 128, S / 128), 256, 0, stream>>>(Ob, wob, out, S, DIM, DIM);
}
